// SWPatchEncoder_60894046323276
// MI455X (gfx1250) — matrix operand order repaired and checked by reading, not compiled, not run on hardware
//
#include <hip/hip_runtime.h>

typedef _Float16 v16h __attribute__((ext_vector_type(16)));
typedef _Float16 v8h  __attribute__((ext_vector_type(8)));
typedef float    v8f  __attribute__((ext_vector_type(8)));

#define STRIDE    8
#define D_MODEL   128
#define T_LEN     3000
#define N_PATCH   375       // (3000 + 8 - 16)/8 + 1
#define MT        24        // ceil(375/16) M-tiles per sequence
#define XWIN      136       // 15*8 + 16 floats per 16-patch tile
#define HRS       136       // h row stride in halves: 68 dwords = 4 mod 64 -> conflict-free

// ws layout (halves):
//   w1_hi[2*128*16]  @ 0        (K padded 16->32, zeros baked for K>=16)
//   w1_lo[2*128*16]  @ 4096
//   w2_hi[8*128*16]  @ 8192     (4 K-chunks x 2 half-chunks)
//   w2_lo[8*128*16]  @ 24576
#define W1HI_OFF  0
#define W1LO_OFF  4096
#define W2HI_OFF  8192
#define W2LO_OFF  24576
#define N_W1      4096
#define N_W2      16384

// ---------------- prologue: split weights to f16 hi/lo in B-fragment layout ----------------
__global__ __launch_bounds__(256)
void split_weights_kernel(const float* __restrict__ W1,
                          const float* __restrict__ W2,
                          _Float16* __restrict__ ws)
{
    const int idx = blockIdx.x * 256 + threadIdx.x;   // 0 .. 20479
    if (idx < N_W1) {
        // W1 (padded to K=32): idx = (h16*128 + n)*16 + j ; K = (j&8)*2 + h16*8 + (j&7)
        const int j   = idx & 15;
        const int n   = (idx >> 4) & 127;
        const int h16 = idx >> 11;
        const int k   = (j & 8) * 2 + h16 * 8 + (j & 7);
        const float v = (k < 16) ? W1[k * D_MODEL + n] : 0.f;
        const _Float16 hi = (_Float16)v;
        const _Float16 lo = (_Float16)(v - (float)hi);
        ws[W1HI_OFF + idx] = hi;
        ws[W1LO_OFF + idx] = lo;
    } else {
        // W2: idx2 = ((c*2 + h16)*128 + n)*16 + j ; K = c*32 + (j&8)*2 + h16*8 + (j&7)
        const int idx2 = idx - N_W1;
        const int j   = idx2 & 15;
        const int n   = (idx2 >> 4) & 127;
        const int g   = idx2 >> 11;        // 0..7
        const int c   = g >> 1;
        const int h16 = g & 1;
        const int k   = c * 32 + (j & 8) * 2 + h16 * 8 + (j & 7);
        const float v = W2[k * D_MODEL + n];
        const _Float16 hi = (_Float16)v;
        const _Float16 lo = (_Float16)(v - (float)hi);
        ws[W2HI_OFF + idx2] = hi;
        ws[W2LO_OFF + idx2] = lo;
    }
}

// ---------------- main kernel ----------------
__global__ __launch_bounds__(256)
void swpatch_wmma_f16split_kernel(const float* __restrict__ x,
                                  const float* __restrict__ b1,
                                  const float* __restrict__ b2,
                                  const _Float16* __restrict__ ws,
                                  float* __restrict__ out)
{
    __shared__ float    lds_x[XWIN];
    __shared__ _Float16 lds_hhi[16 * HRS];
    __shared__ _Float16 lds_hlo[16 * HRS];

    const int mtile = blockIdx.x;        // 0..23
    const int be    = blockIdx.y;        // b*E + e
    const int tid   = threadIdx.x;
    const int lane  = tid & 31;          // wave32
    const int wave  = tid >> 5;          // 0..7 -> N tile
    const int nbase = wave * 16;

    // stage 136-float x window (edge clamp == ReplicationPad1d)
    const long xbase = (long)be * T_LEN;
    if (tid < XWIN) {
        int t = mtile * (16 * STRIDE) + tid;
        if (t > T_LEN - 1) t = T_LEN - 1;
        lds_x[tid] = x[xbase + t];
    }
    __syncthreads();

    const int mrow = lane & 15;          // M (A) / N (B) index in 16x16 tile
    const int h16  = lane >> 4;          // 0 or 1
    const int khw  = h16 * 8;            // A-fragment K base for halves 0..7
    const int ncol = nbase + mrow;

    // ---- GEMM1: patches[16x16(pad32)] @ W1, 3 x WMMA f32_16x16x32_f16 ----
    v16h a_hi, a_lo;
#pragma unroll
    for (int j = 0; j < 8; ++j) {
        const float v = lds_x[mrow * STRIDE + khw + j];   // 8 contiguous floats
        const _Float16 hi = (_Float16)v;
        a_hi[j] = hi;
        a_lo[j] = (_Float16)(v - (float)hi);
        a_hi[j + 8] = (_Float16)0.f;                      // K>=16 pad (B also zero)
        a_lo[j + 8] = (_Float16)0.f;
    }
    const v16h b1hi = *(const v16h*)(ws + W1HI_OFF + (((h16 << 7) + ncol) << 4));
    const v16h b1lo = *(const v16h*)(ws + W1LO_OFF + (((h16 << 7) + ncol) << 4));

    v8f acc = {0.f, 0.f, 0.f, 0.f, 0.f, 0.f, 0.f, 0.f};
    acc = __builtin_amdgcn_wmma_f32_16x16x32_f16(false, a_hi, false, b1hi, (short)0, acc, false, false);
    acc = __builtin_amdgcn_wmma_f32_16x16x32_f16(false, a_hi, false, b1lo, (short)0, acc, false, false);
    acc = __builtin_amdgcn_wmma_f32_16x16x32_f16(false, a_lo, false, b1hi, (short)0, acc, false, false);

    // bias + ReLU, split h to hi/lo halves in LDS
    const float bias1 = b1[ncol];
#pragma unroll
    for (int r = 0; r < 8; ++r) {
        float v = acc[r] + bias1;
        v = v > 0.f ? v : 0.f;
        const int m = r + h16 * 8;
        const _Float16 hhi = (_Float16)v;
        lds_hhi[m * HRS + ncol] = hhi;
        lds_hlo[m * HRS + ncol] = (_Float16)(v - (float)hhi);
    }
    __syncthreads();

    // ---- GEMM2: h[16x128] @ W2, 4 K-chunks x 3 WMMA ----
    v8f acc2 = {0.f, 0.f, 0.f, 0.f, 0.f, 0.f, 0.f, 0.f};
#pragma unroll
    for (int c = 0; c < 4; ++c) {
        const int off0 = mrow * HRS + c * 32 + khw;        // halves 0..7  (16B aligned)
        const int off1 = off0 + 16;                        // halves 8..15
        v16h ah, al;
        ((v8h*)&ah)[0] = *(const v8h*)(lds_hhi + off0);
        ((v8h*)&ah)[1] = *(const v8h*)(lds_hhi + off1);
        ((v8h*)&al)[0] = *(const v8h*)(lds_hlo + off0);
        ((v8h*)&al)[1] = *(const v8h*)(lds_hlo + off1);

        const int bidx = (((c * 2 + h16) << 7) + ncol) << 4;
        const v16h bh = *(const v16h*)(ws + W2HI_OFF + bidx);
        const v16h bl = *(const v16h*)(ws + W2LO_OFF + bidx);

        acc2 = __builtin_amdgcn_wmma_f32_16x16x32_f16(false, ah, false, bh, (short)0, acc2, false, false);
        acc2 = __builtin_amdgcn_wmma_f32_16x16x32_f16(false, ah, false, bl, (short)0, acc2, false, false);
        acc2 = __builtin_amdgcn_wmma_f32_16x16x32_f16(false, al, false, bh, (short)0, acc2, false, false);
    }

    // ---- bias + predicated coalesced store ----
    const float bias2 = b2[ncol];
    const long  obase = (long)be * N_PATCH * D_MODEL;
#pragma unroll
    for (int r = 0; r < 8; ++r) {
        const int m = r + h16 * 8;
        const int p = mtile * 16 + m;
        if (p < N_PATCH) {
            out[obase + (long)p * D_MODEL + ncol] = acc2[r] + bias2;
        }
    }
}

extern "C" void kernel_launch(void* const* d_in, const int* in_sizes, int n_in,
                              void* d_out, int out_size, void* d_ws, size_t ws_size,
                              hipStream_t stream) {
    (void)n_in; (void)out_size; (void)ws_size;
    const float* x  = (const float*)d_in[0];
    const float* W1 = (const float*)d_in[1];
    const float* b1 = (const float*)d_in[2];
    const float* W2 = (const float*)d_in[3];
    const float* b2 = (const float*)d_in[4];
    float*     out  = (float*)d_out;
    _Float16*  ws   = (_Float16*)d_ws;     // needs 80 KB

    // prologue: split weights once (stream-ordered before main kernel)
    split_weights_kernel<<<(N_W1 + N_W2) / 256, 256, 0, stream>>>(W1, W2, ws);

    const int BE = in_sizes[0] / T_LEN;    // B * E = 2048
    dim3 grid(MT, BE, 1);
    dim3 block(256, 1, 1);
    swpatch_wmma_f16split_kernel<<<grid, block, 0, stream>>>(x, b1, b2, ws, out);
}
